// RudderLSTM_45887430591102
// MI455X (gfx1250) — matrix operand order repaired and checked by reading, not compiled, not run on hardware
//
#include <hip/hip_runtime.h>
#include <hip/hip_bf16.h>

#define T_STEPS 2048
#define BATCH   64
#define HID     512
#define NWG     32
#define DPAD    96   // D_IN=67 padded to 3 k-tiles of 32

typedef __attribute__((ext_vector_type(8)))  __bf16 v8bf;
typedef __attribute__((ext_vector_type(16))) __bf16 v16bf;
typedef __attribute__((ext_vector_type(8)))  float  v8f;
typedef __attribute__((ext_vector_type(4)))  unsigned uint32x4;
typedef __attribute__((ext_vector_type(8)))  int      int32x8;
typedef __attribute__((ext_vector_type(4)))  int      int32x4;

// A-matrix fragment (16x32 bf16): lane m=l&15, half=l>>4.
// VGPR0-3 <- K = half*8 .. +7 ; VGPR4-7 <- K = half*8+16 .. +23  (two 16B loads)
__device__ __forceinline__ v16bf load_a16(const __bf16* p) {
  v8bf lo = *(const v8bf*)(p);
  v8bf hi = *(const v8bf*)(p + 16);
  return __builtin_shufflevector(lo, hi, 0,1,2,3,4,5,6,7,8,9,10,11,12,13,14,15);
}
// B-matrix fragment (32x16 bf16): lane n=l&15; K = half*8..+7 and half*8+16..+23 (two 16B loads)
__device__ __forceinline__ v16bf load_b16(const __bf16* p) {
  v8bf lo = *(const v8bf*)(p);
  v8bf hi = *(const v8bf*)(p + 16);
  return __builtin_shufflevector(lo, hi, 0,1,2,3,4,5,6,7,8,9,10,11,12,13,14,15);
}
__device__ __forceinline__ v8f wmma_bf16(v16bf a, v16bf b, v8f c) {
  return __builtin_amdgcn_wmma_f32_16x16x32_bf16(false, a, false, b, (short)0, c, false, false);
}
__device__ __forceinline__ float sigm(float x)      { return 1.f / (1.f + __expf(-x)); }
__device__ __forceinline__ float tanh_fast(float x) { float e = __expf(-2.f * x); return (1.f - e) / (1.f + e); }

// ---------------------------------------------------------------------------
// TDM: DMA a contiguous `bytes` (multiple of 8, <= 512KB) block global -> LDS.
// D#: count=1, type=2(image), gather off; data_size=8B; 1-D tile.
// ---------------------------------------------------------------------------
#if defined(__has_builtin)
#if __has_builtin(__builtin_amdgcn_tensor_load_to_lds) && __has_builtin(__builtin_amdgcn_s_wait_tensorcnt)
#define HAVE_TDM 1
#endif
#endif

__device__ __forceinline__ void tdm_load_1d(const void* gptr, unsigned lds_off, unsigned bytes) {
#ifdef HAVE_TDM
  unsigned long long va = (unsigned long long)(uintptr_t)gptr;
  unsigned n8 = bytes >> 3;                    // elements of 8 bytes
  uint32x4 g0;
  g0[0] = 1u;                                  // count=1, is_restore=0, gather off
  g0[1] = lds_off;                             // lds_addr [63:32]
  g0[2] = (unsigned)(va & 0xFFFFFFFFu);        // global_addr [95:64]
  g0[3] = (unsigned)((va >> 32) & 0x1FFFFFFu)  // global_addr [120:96]
        | 0x80000000u;                         // type=2 -> bits[127:126]=10b
  int32x8 g1;
  g1[0] = 0x00030000;                          // workgroup_mask=0, data_size=3 (8B)
  g1[1] = (int)((n8 & 0xFFFFu) << 16);         // tensor_dim0[15:0] @ bits63:48
  g1[2] = (int)(((n8 >> 16) & 0xFFFFu))        // tensor_dim0[31:16] @ bits79:64
        | (1 << 16);                           // tensor_dim1 = 1    @ bits95:80
  g1[3] = (int)((n8 & 0xFFFFu) << 16);         // tile_dim0 @ bits127:112
  g1[4] = 1;                                   // tile_dim1 = 1 @ bits143:128
  g1[5] = (int)n8;                             // tensor_dim0_stride[31:0]
  g1[6] = 0;                                   // dim0_stride[47:32], dim1_stride lo
  g1[7] = 0;
  int32x4 gz = {0, 0, 0, 0};
#if __clang_major__ >= 23
  int32x8 gz8 = {0, 0, 0, 0, 0, 0, 0, 0};
  __builtin_amdgcn_tensor_load_to_lds(g0, g1, gz, gz, gz8, 0);
#else
  __builtin_amdgcn_tensor_load_to_lds(g0, g1, gz, gz, 0);
#endif
#endif
}

// ---------------------------------------------------------------------------
// Prep: bf16 weight transposes (N-major, K-contiguous), zero h0 row + counter
// ---------------------------------------------------------------------------
__global__ void prep_kernel(const float* __restrict__ Wh, const float* __restrict__ Wi,
                            const float* __restrict__ W1,
                            __bf16* __restrict__ wt_h, __bf16* __restrict__ wt_i,
                            __bf16* __restrict__ wt1, __bf16* __restrict__ hist,
                            unsigned* __restrict__ ctr) {
  size_t i = (size_t)blockIdx.x * blockDim.x + threadIdx.x;
  size_t stride = (size_t)gridDim.x * blockDim.x;
  if (i == 0) *ctr = 0u;
  for (size_t j = i; j < (size_t)2048 * HID; j += stride) {      // W_h: [n][k]
    size_t n = j >> 9, k = j & 511;
    wt_h[j] = (__bf16)Wh[k * 2048 + n];
  }
  for (size_t j = i; j < (size_t)2048 * DPAD; j += stride) {     // W_i: [n][k], pad K>=67 with 0
    size_t n = j / DPAD, k = j % DPAD;
    wt_i[j] = (k < 67) ? (__bf16)Wi[k * 2048 + n] : (__bf16)0.f;
  }
  for (size_t j = i; j < (size_t)256 * HID; j += stride) {       // W1: [n][k]
    size_t n = j >> 9, k = j & 511;
    wt1[j] = (__bf16)W1[k * 256 + n];
  }
  for (size_t j = i; j < (size_t)BATCH * HID; j += stride)       // h_{-1} = 0
    hist[j] = (__bf16)0.f;
}

// ---------------------------------------------------------------------------
// Persistent LSTM: 32 WGs x 128 threads; WG owns 16 hidden cols (x4 gates).
// Wave w handles batch rows [16w,16w+16). c-state stays in registers.
// ---------------------------------------------------------------------------
__global__ void __launch_bounds__(128, 1)
lstm_kernel(const int* __restrict__ boss_anim, const int* __restrict__ hero_anim,
            const float* __restrict__ cont, const float* __restrict__ btab,
            const float* __restrict__ htab, const __bf16* __restrict__ wt_i,
            const __bf16* __restrict__ wt_h, const float* __restrict__ b_lstm,
            __bf16* __restrict__ hist, unsigned* __restrict__ ctr) {
  __shared__ __align__(16) __bf16 xs[BATCH * DPAD];   // 12 KB: x_t tile, K-padded

  const int tid   = threadIdx.x;
  const int wg    = blockIdx.x;        // owns hidden cols [wg*16, wg*16+16)
  const int lane  = tid & 31;
  const int wave  = tid >> 5;          // mTile (batch rows 16*wave..+15)
  const int kh    = lane >> 4;         // "half" in WMMA layouts
  const int col   = lane & 15;
  const int hbase = wg * 16;
  const int arow  = wave * 16 + col;   // A-matrix row this lane loads

  // zero the K-pad columns once (gather never touches c >= 67)
  for (int i = tid; i < BATCH * DPAD; i += 128)
    if ((i % DPAD) >= 67) xs[i] = (__bf16)0.f;

  // per-lane gate biases (i,f,g,o), fixed for the whole scan
  const float bi = b_lstm[0 * HID + hbase + col];
  const float bf = b_lstm[1 * HID + hbase + col];
  const float bg = b_lstm[2 * HID + hbase + col];
  const float bo = b_lstm[3 * HID + hbase + col];

  float cst[8];
  #pragma unroll
  for (int r = 0; r < 8; ++r) cst[r] = 0.f;

  const int grow = tid >> 1;           // gather: 2 threads per batch row
  const int gc0  = (tid & 1) * 48;

  for (int t = 0; t < T_STEPS; ++t) {
    // ---- fused embedding gather into LDS (bf16) ----
    {
      int bidx = boss_anim[grow * T_STEPS + t];
      int hidx = hero_anim[grow * T_STEPS + t];
      bidx = bidx < 0 ? 0 : (bidx > 49 ? 49 : bidx);
      hidx = hidx < 0 ? 0 : (hidx > 199 ? 199 : hidx);
      for (int c = gc0; c < gc0 + 48 && c < 67; ++c) {
        float v;
        if (c < 32)      v = btab[bidx * 32 + c];
        else if (c < 64) v = htab[hidx * 32 + (c - 32)];
        else             v = cont[((size_t)grow * T_STEPS + t) * 3 + (c - 64)];
        xs[grow * DPAD + c] = (__bf16)v;
      }
    }
    // prefetch next step's index/continuous rows (global_prefetch_b8)
    if (t + 1 < T_STEPS) {
      __builtin_prefetch(&boss_anim[grow * T_STEPS + t + 1], 0, 0);
      __builtin_prefetch(&hero_anim[grow * T_STEPS + t + 1], 0, 0);
      __builtin_prefetch(&cont[((size_t)grow * T_STEPS + t + 1) * 3], 0, 0);
    }
    __syncthreads();

    v8f acc[4] = {v8f{}, v8f{}, v8f{}, v8f{}};   // nt = gate index (i,f,g,o)

    // ---- input projection: x_t(64x96) @ W_i-slice ----
    #pragma unroll
    for (int kt = 0; kt < 3; ++kt) {
      v16bf a = load_a16(&xs[arow * DPAD + kt * 32 + kh * 8]);
      #pragma unroll
      for (int nt = 0; nt < 4; ++nt) {
        v16bf b = load_b16(wt_i + (size_t)(nt * HID + hbase + col) * DPAD + kt * 32 + kh * 8);
        acc[nt] = wmma_bf16(a, b, acc[nt]);
      }
    }

    // ---- recurrent: h_t(64x512) @ W_h-slice ----
    const __bf16* hp = hist + (size_t)t * (BATCH * HID);
    #pragma unroll 4
    for (int kt = 0; kt < 16; ++kt) {
      v16bf a = load_a16(hp + (size_t)arow * HID + kt * 32 + kh * 8);
      #pragma unroll
      for (int nt = 0; nt < 4; ++nt) {
        v16bf b = load_b16(wt_h + (size_t)(nt * HID + hbase + col) * HID + kt * 32 + kh * 8);
        acc[nt] = wmma_bf16(a, b, acc[nt]);
      }
    }

    // ---- gate math + state update; D layout: (m = 16*wave + r + 8*kh, n = col) ----
    __bf16* hn = hist + (size_t)(t + 1) * (BATCH * HID);
    #pragma unroll
    for (int r = 0; r < 8; ++r) {
      float iv = sigm(acc[0][r] + bi);
      float fv = sigm(acc[1][r] + bf);
      float gv = tanh_fast(acc[2][r] + bg);
      float ov = sigm(acc[3][r] + bo);
      float c  = fv * cst[r] + iv * gv;
      cst[r]   = c;
      float h  = ov * tanh_fast(c);
      int m    = wave * 16 + r + 8 * kh;
      hn[(size_t)m * HID + hbase + col] = (__bf16)h;
    }

    // ---- device-wide step barrier (monotonic counter) ----
    __syncthreads();
    if (tid == 0) {
      __threadfence();
      atomicAdd(ctr, 1u);
      const unsigned target = (unsigned)(t + 1) * NWG;
      while (__hip_atomic_load(ctr, __ATOMIC_RELAXED, __HIP_MEMORY_SCOPE_AGENT) < target)
        __builtin_amdgcn_s_sleep(1);
    }
    __syncthreads();
    __threadfence();
  }
}

// ---------------------------------------------------------------------------
// Head: z = relu(hs @ W1 + b1); preds = z @ W2 + b2. WG = 64 rows of B*T.
// hs tile (64x512 bf16 = 64KB) is staged to LDS once via the Tensor Data
// Mover, then A-fragments come from LDS across all 4 N-groups.
// ---------------------------------------------------------------------------
__global__ void __launch_bounds__(128, 1)
head_kernel(const __bf16* __restrict__ hist, const __bf16* __restrict__ wt1,
            const float* __restrict__ b1, const float* __restrict__ W2,
            const float* __restrict__ b2, float* __restrict__ out) {
  __shared__ __align__(16) __bf16 hstile[BATCH * HID];   // 64 KB

  const int tid  = threadIdx.x;
  const int lane = tid & 31;
  const int wave = tid >> 5;
  const int kh   = lane >> 4;
  const int col  = lane & 15;
  const size_t wgq = (size_t)blockIdx.x * 64;            // first row of this WG
  // hs row q = t*64+b lives at hist row q+64 (rows 0..63 hold h_{-1}=0)
  const __bf16* gsrc = hist + (wgq + 64) * HID;

#ifdef HAVE_TDM
  if (wave == 0) {   // one TDM issue per WG (EXEC ignored by TDM)
    tdm_load_1d(gsrc, (unsigned)(uintptr_t)&hstile[0], BATCH * HID * 2);
    __builtin_amdgcn_s_wait_tensorcnt(0);
  }
#else
  for (int i = tid * 8; i < BATCH * HID; i += 128 * 8)
    *(v8bf*)&hstile[i] = *(const v8bf*)&gsrc[i];
#endif
  __syncthreads();

  const __bf16* Ap = hstile + (size_t)(wave * 16 + col) * HID;   // lane's A row

  float partial[8];
  #pragma unroll
  for (int r = 0; r < 8; ++r) partial[r] = 0.f;
  const float b2v = b2[0];

  #pragma unroll
  for (int grp = 0; grp < 4; ++grp) {            // 4 groups of 64 N-cols
    v8f acc[4] = {v8f{}, v8f{}, v8f{}, v8f{}};
    #pragma unroll 4
    for (int kt = 0; kt < 16; ++kt) {
      v16bf a = load_a16(Ap + kt * 32 + kh * 8);
      #pragma unroll
      for (int nt = 0; nt < 4; ++nt) {
        int n = grp * 64 + nt * 16 + col;
        v16bf bm = load_b16(wt1 + (size_t)n * HID + kt * 32 + kh * 8);
        acc[nt] = wmma_bf16(a, bm, acc[nt]);
      }
    }
    #pragma unroll
    for (int nt = 0; nt < 4; ++nt) {
      int n = grp * 64 + nt * 16 + col;
      float b1v = b1[n], w2v = W2[n];
      #pragma unroll
      for (int r = 0; r < 8; ++r) {
        float z = fmaxf(acc[nt][r] + b1v, 0.f);  // relu
        partial[r] += z * w2v;                   // fold W2 dot
      }
    }
  }

  // reduce over the 16 N-lanes (xor keeps the kh bit intact)
  #pragma unroll
  for (int r = 0; r < 8; ++r) {
    float v = partial[r];
    v += __shfl_xor(v, 1, 32);
    v += __shfl_xor(v, 2, 32);
    v += __shfl_xor(v, 4, 32);
    v += __shfl_xor(v, 8, 32);
    if (col == 0) {
      size_t q = wgq + (size_t)(wave * 16 + r + 8 * kh);
      int t = (int)(q >> 6), b = (int)(q & 63);
      out[(size_t)b * T_STEPS + t] = v + b2v;
    }
  }
}

// ---------------------------------------------------------------------------
extern "C" void kernel_launch(void* const* d_in, const int* in_sizes, int n_in,
                              void* d_out, int out_size, void* d_ws, size_t ws_size,
                              hipStream_t stream) {
  const int*   boss_anim = (const int*)d_in[0];
  const int*   hero_anim = (const int*)d_in[1];
  const float* cont      = (const float*)d_in[2];
  const float* btab      = (const float*)d_in[3];
  const float* htab      = (const float*)d_in[4];
  const float* Wi        = (const float*)d_in[5];
  const float* Wh        = (const float*)d_in[6];
  const float* b_lstm    = (const float*)d_in[7];
  const float* W1        = (const float*)d_in[8];
  const float* b1        = (const float*)d_in[9];
  const float* W2        = (const float*)d_in[10];
  const float* b2        = (const float*)d_in[11];
  float* out = (float*)d_out;

  char* ws = (char*)d_ws;
  unsigned* ctr = (unsigned*)ws;                                    // 256B reserved
  __bf16* hist  = (__bf16*)(ws + 256);                              // (T+1)*64*512
  size_t histBytes = (size_t)(T_STEPS + 1) * BATCH * HID * sizeof(__bf16);
  __bf16* wt_h = (__bf16*)(ws + 256 + histBytes);                   // 2048*512
  __bf16* wt_i = wt_h + (size_t)2048 * HID;                         // 2048*96
  __bf16* wt1  = wt_i + (size_t)2048 * DPAD;                        // 256*512

  prep_kernel<<<2048, 256, 0, stream>>>(Wh, Wi, W1, wt_h, wt_i, wt1, hist, ctr);
  lstm_kernel<<<NWG, 128, 0, stream>>>(boss_anim, hero_anim, cont, btab, htab,
                                       wt_i, wt_h, b_lstm, hist, ctr);
  head_kernel<<<2048, 128, 0, stream>>>(hist, wt1, b1, W2, b2, out);
}
